// LiquidCell_43817256353921
// MI455X (gfx1250) — matrix operand order repaired and checked by reading, not compiled, not run on hardware
//
#include <hip/hip_runtime.h>
#include <hip/hip_bf16.h>

// ---------------------------------------------------------------------------
// LiquidCell on MI455X (gfx1250, wave32, WMMA).
//
// Phase 0: x (fp32, 128MB) -> x_bf16 (64MB, one streaming pass)
//          w_in, w_rec -> bf16 transposed (N-major) so every per-lane WMMA
//          B fragment is two contiguous 16-byte chunks.
// Phase 1: x_proj = x @ w_in + b  (M=16384, K=2048, N=512) via
//          v_wmma_f32_16x16x32_bf16; A and B fragments loaded straight from
//          global bf16 (L2-resident), no LDS/barriers in the K loop.
//          Wave tile 32x64 -> 8 WMMA per 12 b128 loads per K-chunk.
// Phase 2: persistent scan. Recurrence is independent per batch row, so 8
//          workgroups each own 16 rows of h and keep it in LDS (fp32+bf16)
//          across all 128 timesteps. No global sync, no per-step launches.
// ---------------------------------------------------------------------------

typedef __bf16 bf16;
typedef __attribute__((ext_vector_type(16))) __bf16 v16bf;
typedef __attribute__((ext_vector_type(8)))  float  v8f;
typedef __attribute__((ext_vector_type(4)))  unsigned int u32x4;

#define B_DIM 128
#define T_DIM 128
#define D_DIM 2048
#define U_DIM 512

// Load a 16-element bf16 WMMA fragment as two contiguous 16-byte chunks.
__device__ __forceinline__ v16bf load16(const bf16* p0, const bf16* p1) {
  union { v16bf v; u32x4 q[2]; } f;
  f.q[0] = *reinterpret_cast<const u32x4*>(p0);
  f.q[1] = *reinterpret_cast<const u32x4*>(p1);
  return f.v;
}

__device__ __forceinline__ v8f wmma_bf16(v16bf a, v16bf b, v8f c) {
  return __builtin_amdgcn_wmma_f32_16x16x32_bf16(
      /*neg_a=*/false, a, /*neg_b=*/false, b,
      /*c_mod=*/(short)0, c, /*reuse_a=*/false, /*reuse_b=*/false);
}

// ---------------------------------------------------------------------------
// fp32 -> bf16 streaming conversion (4 elements/thread, coalesced float4)
// ---------------------------------------------------------------------------
__global__ void __launch_bounds__(256)
convert_bf16(const float* __restrict__ src, bf16* __restrict__ dst, int n4) {
  int i = blockIdx.x * 256 + threadIdx.x;
  if (i < n4) {
    float4 v = reinterpret_cast<const float4*>(src)[i];
    bf16* d = dst + (size_t)i * 4;
    d[0] = (bf16)v.x; d[1] = (bf16)v.y; d[2] = (bf16)v.z; d[3] = (bf16)v.w;
  }
}

// ---------------------------------------------------------------------------
// src[rows][cols] fp32  ->  dst[cols][rows] bf16   (LDS-tiled, conflict-free)
// block (32,8), grid (cols/32, rows/32)
// ---------------------------------------------------------------------------
__global__ void __launch_bounds__(256)
transpose_to_bf16(const float* __restrict__ src, bf16* __restrict__ dst,
                  int rows, int cols) {
  __shared__ float tile[32][33];
  const int r0 = blockIdx.y * 32, c0 = blockIdx.x * 32;
  const int tx = threadIdx.x, ty = threadIdx.y;
  #pragma unroll
  for (int i = 0; i < 32; i += 8)
    tile[ty + i][tx] = src[(size_t)(r0 + ty + i) * cols + c0 + tx];
  __syncthreads();
  #pragma unroll
  for (int i = 0; i < 32; i += 8)
    dst[(size_t)(c0 + ty + i) * rows + r0 + tx] = (bf16)tile[tx][ty + i];
}

// ---------------------------------------------------------------------------
// Phase 1 GEMM: xp[M][512] = x_bf16[M][2048] @ w_in + bias, M = B*T = 16384.
// Block tile 128(M) x 128(N); 8 waves as 4(M) x 2(N); wave = 32 x 64.
// grid (N/128=4, M/128=128), block 256. No LDS: fragments direct from L2.
// ---------------------------------------------------------------------------
__global__ void __launch_bounds__(256)
xproj_gemm(const bf16* __restrict__ xb, const bf16* __restrict__ wT,
           const float* __restrict__ bias, float* __restrict__ xp) {
  const int tid   = threadIdx.x;
  const int wave  = tid >> 5;
  const int lane  = tid & 31;
  const int m16   = lane & 15;
  const int ahalf = lane >> 4;        // ISA 7.12.2: lane half selects K group
  const int akoff = ahalf * 8;        // A: K {0-7,16-23} vs {8-15,24-31}
  const int bkoff = ahalf * 16;       // unused; B loads use akoff, same K as A
  const int wm    = wave >> 1;        // 0..3 : M subtile
  const int wn    = wave & 1;         // 0..1 : N subtile

  const size_t rBase = (size_t)blockIdx.y * 128 + (size_t)wm * 32;
  const int    cBase = blockIdx.x * 128 + wn * 64;

  const bf16* arow0 = xb + (rBase + m16) * D_DIM + akoff;        // rows 0-15
  const bf16* arow1 = xb + (rBase + 16 + m16) * D_DIM + akoff;   // rows 16-31

  v8f acc[2][4] = {};

  for (int kb = 0; kb < D_DIM; kb += 32) {
    __builtin_prefetch(arow0 + kb + 512, 0, 1);   // 16 K-tiles (512 elements) ahead
    __builtin_prefetch(arow1 + kb + 512, 0, 1);
    v16bf a0 = load16(arow0 + kb, arow0 + kb + 16);
    v16bf a1 = load16(arow1 + kb, arow1 + kb + 16);
    #pragma unroll
    for (int t = 0; t < 4; ++t) {
      const bf16* bp = wT + (size_t)(cBase + t * 16 + m16) * D_DIM + kb + akoff;
      v16bf bfrag = load16(bp, bp + 16);
      acc[0][t] = wmma_bf16(a0, bfrag, acc[0][t]);
      acc[1][t] = wmma_bf16(a1, bfrag, acc[1][t]);
    }
  }

  // C layout (ISA): lanes 0-15 -> rows r, lanes 16-31 -> rows r+8.
  #pragma unroll
  for (int t = 0; t < 4; ++t) {
    int col = cBase + t * 16 + m16;
    float bv = bias[col];
    #pragma unroll
    for (int r = 0; r < 8; ++r) {
      size_t row = rBase + (ahalf ? r + 8 : r);
      xp[row * U_DIM + col]        = acc[0][t][r] + bv;
      xp[(row + 16) * U_DIM + col] = acc[1][t][r] + bv;
    }
  }
}

// ---------------------------------------------------------------------------
// Phase 2: persistent scan. grid = 8 WGs, each owns 16 batch rows for all
// 128 timesteps. h lives in LDS (fp32 for elementwise, bf16 for WMMA A).
// Per step: GEMM 16x512x512 (8 waves x 64 cols, 4 acc each, K-chunks of 32),
// liquid gating, per-row LayerNorm via shfl_xor + two barriers.
// ---------------------------------------------------------------------------
__global__ void __launch_bounds__(256)
liquid_scan(const float* __restrict__ xp, const bf16* __restrict__ wrecT,
            const float* __restrict__ tau, const float* __restrict__ gamma,
            const float* __restrict__ beta, float* __restrict__ out) {
  __shared__ bf16  hb[16][U_DIM];   // 16 KB  (WMMA A operand)
  __shared__ float hf[16][U_DIM];   // 32 KB  (fp32 state)
  __shared__ float ns[16][U_DIM];   // 32 KB  (pre-LN new_state)

  const int tid   = threadIdx.x;
  const int wave  = tid >> 5;
  const int lane  = tid & 31;
  const int m16   = lane & 15;
  const int ahalf = lane >> 4;
  const int akoff = ahalf * 8;
  const int bkoff = ahalf * 16;
  const int b0    = blockIdx.x * 16;   // this WG's batch rows
  const int cw    = wave * 64;         // this wave's column base

  for (int i = tid; i < 16 * U_DIM; i += 256) {
    (&hb[0][0])[i] = (bf16)0.0f;
    (&hf[0][0])[i] = 0.0f;
  }
  __syncthreads();

  for (int t = 0; t < T_DIM; ++t) {
    // ---- GEMM: acc = h(16x512) @ w_rec(512 x [cw..cw+63]) ----
    v8f acc[4] = {};
    for (int kb = 0; kb < U_DIM; kb += 32) {
      v16bf a = load16(&hb[m16][kb + akoff], &hb[m16][kb + akoff + 16]);
      #pragma unroll
      for (int n = 0; n < 4; ++n) {
        const bf16* bp =
            wrecT + (size_t)(cw + n * 16 + m16) * U_DIM + kb + akoff;
        v16bf bfrag = load16(bp, bp + 16);
        acc[n] = wmma_bf16(a, bfrag, acc[n]);
      }
    }

    // ---- liquid gating: new_state = h + (base - h)*(tau*sig(base)+eps) ----
    #pragma unroll
    for (int n = 0; n < 4; ++n) {
      int col = cw + n * 16 + m16;
      float tv = tau[col];
      #pragma unroll
      for (int r = 0; r < 8; ++r) {
        int lr = ahalf ? r + 8 : r;
        float base = acc[n][r] +
            xp[((size_t)(b0 + lr) * T_DIM + t) * U_DIM + col];
        float h0 = hf[lr][col];
        float sg = 1.0f / (1.0f + __expf(-base));
        float lt = tv * sg + 1e-7f;
        ns[lr][col] = h0 + (base - h0) * lt;
      }
    }
    __syncthreads();

    // ---- LayerNorm: 16 lanes per row, 32 cols per lane ----
    {
      int lr = tid >> 4, seg = tid & 15;
      float s = 0.0f, ss = 0.0f;
      #pragma unroll
      for (int i = 0; i < 32; ++i) {
        float v = ns[lr][seg * 32 + i];
        s += v; ss += v * v;
      }
      #pragma unroll
      for (int off = 1; off < 16; off <<= 1) {   // reduce within 16-lane group
        s  += __shfl_xor(s,  off, 32);
        ss += __shfl_xor(ss, off, 32);
      }
      float mean = s * (1.0f / 512.0f);
      float var  = ss * (1.0f / 512.0f) - mean * mean;
      float inv  = rsqrtf(var + 1e-3f);
      #pragma unroll
      for (int i = 0; i < 32; ++i) {
        int c = seg * 32 + i;
        float hn = gamma[c] * (ns[lr][c] - mean) * inv + beta[c];
        hf[lr][c] = hn;
        hb[lr][c] = (bf16)hn;
      }
    }
    __syncthreads();   // hb/hf stable before next step's GEMM
  }

  for (int i = tid; i < 16 * U_DIM; i += 256) {
    int lr = i >> 9, c = i & 511;
    out[(size_t)(b0 + lr) * U_DIM + c] = hf[lr][c];
  }
}

// ---------------------------------------------------------------------------
extern "C" void kernel_launch(void* const* d_in, const int* in_sizes, int n_in,
                              void* d_out, int out_size, void* d_ws, size_t ws_size,
                              hipStream_t stream) {
  const float* x     = (const float*)d_in[0];  // [128,128,2048]
  const float* w_in  = (const float*)d_in[1];  // [2048,512]
  const float* w_rec = (const float*)d_in[2];  // [512,512]
  const float* bias  = (const float*)d_in[3];  // [512]
  const float* tau   = (const float*)d_in[4];  // [512]
  const float* gamma = (const float*)d_in[5];  // [512]
  const float* beta  = (const float*)d_in[6];  // [512]
  float* out = (float*)d_out;                  // [128,512]

  // Workspace layout
  const size_t M = (size_t)B_DIM * T_DIM;                    // 16384
  float* xp     = (float*)d_ws;                              // 32 MB
  bf16*  w_inT  = (bf16*)((char*)d_ws + M * U_DIM * sizeof(float));
  bf16*  w_recT = w_inT + (size_t)U_DIM * D_DIM;             // +2 MB
  bf16*  xb     = w_recT + (size_t)U_DIM * U_DIM;            // +512 KB, 64 MB

  // Phase 0: precision conversion / weight transposes
  {
    int n4 = (int)(M * D_DIM / 4);                           // 8388608
    convert_bf16<<<(n4 + 255) / 256, 256, 0, stream>>>(x, xb, n4);
  }
  transpose_to_bf16<<<dim3(U_DIM / 32, D_DIM / 32), dim3(32, 8), 0, stream>>>(
      w_in, w_inT, D_DIM, U_DIM);
  transpose_to_bf16<<<dim3(U_DIM / 32, U_DIM / 32), dim3(32, 8), 0, stream>>>(
      w_rec, w_recT, U_DIM, U_DIM);

  // Phase 1: x_proj = x @ w_in + b
  xproj_gemm<<<dim3(U_DIM / 128, M / 128), 256, 0, stream>>>(xb, w_inT, bias, xp);

  // Phase 2: 128-step persistent scan (8 WGs x 16 batch rows)
  liquid_scan<<<B_DIM / 16, 256, 0, stream>>>(xp, w_recT, tau, gamma, beta, out);

  (void)in_sizes; (void)n_in; (void)out_size; (void)ws_size;
}
